// ShakespeareLSTM_75479755259994
// MI455X (gfx1250) — compile-verified, matrix operand order checked by reading, not run on hardware
//
#include <hip/hip_runtime.h>

typedef __attribute__((ext_vector_type(16))) _Float16 v16h;
typedef __attribute__((ext_vector_type(8)))  float    v8f;

// Problem constants
constexpr int Vn = 80;      // vocab
constexpr int En = 8;       // embed dim
constexpr int Hn = 256;     // hidden
constexpr int Bn = 2048;    // batch
constexpr int Sn = 256;     // seq len
constexpr int G4H = 1024;   // 4*H gate width

// Tiling
constexpr int TILE_B = 16;          // WMMA M
constexpr int NT_TOT = G4H / 16;    // 64 N-tiles of gate matrix
constexpr int WAVES = 8;
constexpr int NT_PER_WAVE = NT_TOT / WAVES;   // 8
constexpr int KC0 = 9;    // layer0 K chunks: 1 (padded emb) + 8 (h0)
constexpr int KC1 = 16;   // layer1 K chunks: 8 (h0) + 8 (h1)
constexpr int KCFC = 8;   // fc K chunks
constexpr int NT_FC = Vn / 16;  // 5

// Pre-swizzled weight fragment sizes (in _Float16 elements).
// Fragment addressing: ((nt*KC + kc)*32 + lane)*16 + h  -> one v16h per lane.
constexpr int F0_ELEMS  = NT_TOT * KC0 * 512;
constexpr int F1_ELEMS  = NT_TOT * KC1 * 512;
constexpr int FFC_ELEMS = NT_FC * KCFC * 512;
constexpr int EMB_ELEMS = Vn * En;            // f16 embedding table copy

// Output layout (f32): logits[2048][80], then h_fin[2][2048][256], then c_fin[2][2048][256]
constexpr int HBASE = Bn * Vn;                 // 163840
constexpr int HSTRIDE = Bn * Hn;               // 524288
constexpr int CBASE = HBASE + 2 * HSTRIDE;     // 1212416

// K index inside a 32-wide chunk for half h (0..15) given lane-half hi (0/1),
// per CDNA5 16-bit WMMA operand layout.
__device__ __forceinline__ int kk_of(int h, int hi) {
    return ((h >> 3) << 4) + hi * 8 + (h & 7);
}

// Gate-interleaved N permutation: permuted column p (0..1023) -> original gate
// row n. Wave w = p>>7 owns columns q = p&127 = gate*32 + u, covering all four
// gates of hidden units j = w*32 + u -> gate math never leaves registers.
__device__ __forceinline__ int mapN(int p) {
    const int q = p & 127;
    return (q >> 5) * Hn + (p >> 7) * 32 + (q & 31);
}

__device__ __forceinline__ float sigf(float x) {
    return __builtin_amdgcn_rcpf(1.f + __expf(-x));
}
__device__ __forceinline__ float tanh_fast(float x) {
    return 1.f - 2.f * __builtin_amdgcn_rcpf(__expf(2.f * x) + 1.f);
}

// ---------------------------------------------------------------------------
// Kernel 1: f32 weights -> f16 WMMA B-fragments (gate-permuted for the LSTM
// layers, natural for FC) + f16 copy of the embedding table, all in workspace.
// ---------------------------------------------------------------------------
__global__ void lstm_prep_weights(const float* __restrict__ Wih0,
                                  const float* __restrict__ Whh0,
                                  const float* __restrict__ Wih1,
                                  const float* __restrict__ Whh1,
                                  const float* __restrict__ fcw,
                                  const float* __restrict__ emb,
                                  _Float16* __restrict__ ws) {
    const long total = (long)F0_ELEMS + F1_ELEMS + FFC_ELEMS + EMB_ELEMS;
    for (long idx = (long)blockIdx.x * blockDim.x + threadIdx.x; idx < total;
         idx += (long)gridDim.x * blockDim.x) {
        long i = idx;
        int which, KC = KCFC;
        if (i < F0_ELEMS)                            { which = 0; KC = KC0; }
        else if (i < (long)F0_ELEMS + F1_ELEMS)      { which = 1; KC = KC1; i -= F0_ELEMS; }
        else if (i < (long)F0_ELEMS + F1_ELEMS + FFC_ELEMS)
                                                     { which = 2; i -= (long)F0_ELEMS + F1_ELEMS; }
        else { ws[idx] = (_Float16)emb[i - ((long)F0_ELEMS + F1_ELEMS + FFC_ELEMS)]; continue; }

        const int h    = (int)(i & 15);
        const int lane = (int)((i >> 4) & 31);
        const long rest = i >> 9;                 // nt*KC + kc
        const int kc = (int)(rest % KC);
        const int nt = (int)(rest / KC);
        const int p  = nt * 16 + (lane & 15);     // output column (permuted for LSTM)
        const int hi = lane >> 4;
        const int kk = kk_of(h, hi);              // 0..31 within chunk

        float val = 0.f;
        if (which == 0) {
            const int n = mapN(p);
            if (kc == 0) { if (kk < En) val = Wih0[n * En + kk]; }
            else         { val = Whh0[n * Hn + (kc - 1) * 32 + kk]; }
        } else if (which == 1) {
            const int n = mapN(p);
            if (kc < 8)  { val = Wih1[n * Hn + kc * 32 + kk]; }
            else         { val = Whh1[n * Hn + (kc - 8) * 32 + kk]; }
        } else {
            val = fcw[p * Hn + kc * 32 + kk];     // natural order, p < 80
        }
        ws[idx] = (_Float16)val;
    }
}

// ---------------------------------------------------------------------------
// Kernel 2: persistent per-batch-tile LSTM. One WG = 16 batch rows, 8 waves,
// 256 timesteps x 2 layers; gates in registers; f16 h states in LDS;
// embeddings arrive via async global->LDS copies; FC head via WMMA.
// ---------------------------------------------------------------------------
__global__ __launch_bounds__(256)
void lstm_main(const int*   __restrict__ x,
               const float* __restrict__ bih0, const float* __restrict__ bhh0,
               const float* __restrict__ bih1, const float* __restrict__ bhh1,
               const float* __restrict__ fcb,
               const _Float16* __restrict__ frag0,
               const _Float16* __restrict__ frag1,
               const _Float16* __restrict__ fragfc,
               const _Float16* __restrict__ emb16,
               float* __restrict__ out) {
    __shared__ _Float16 h0h[TILE_B][Hn];      // 8 KB
    __shared__ _Float16 h1h[TILE_B][Hn];      // 8 KB
    __shared__ _Float16 eA[TILE_B][32];       // 1 KB (padded emb chunk)

    const int tid  = threadIdx.x;
    const int lane = tid & 31;
    const int wv   = tid >> 5;
    const int r0   = blockIdx.x * TILE_B;
    const int m    = lane & 15;   // N-col within 16-tile / A-row
    const int hi   = lane >> 4;

    // per-lane fragment bases: ntl stride becomes an immediate offset
    const _Float16* f0lane = frag0 + (wv * NT_PER_WAVE * KC0 * 32 + lane) * 16;
    const _Float16* f1lane = frag1 + (wv * NT_PER_WAVE * KC1 * 32 + lane) * 16;

    // ---- per-lane combined biases for this wave's 8 output tiles ----
    float bias0r[NT_PER_WAVE], bias1r[NT_PER_WAVE];
    #pragma unroll
    for (int ntl = 0; ntl < NT_PER_WAVE; ++ntl) {
        const int n = mapN((wv * NT_PER_WAVE + ntl) * 16 + m);
        bias0r[ntl] = bih0[n] + bhh0[n];
        bias1r[ntl] = bih1[n] + bhh1[n];
    }

    for (int i = tid; i < TILE_B * Hn; i += 256) {
        (&h0h[0][0])[i] = (_Float16)0.f;
        (&h1h[0][0])[i] = (_Float16)0.f;
    }
    for (int i = tid; i < TILE_B * 32; i += 256) (&eA[0][0])[i] = (_Float16)0.f;

    // c state in registers: c?r[s*8+v] <-> (row = v+hi*8, j = wv*32 + s*16 + m)
    float c0r[16], c1r[16];
    #pragma unroll
    for (int k = 0; k < 16; ++k) { c0r[k] = 0.f; c1r[k] = 0.f; }
    const int eb  = tid >> 4;           // writer mapping for h_fin
    const int ej0 = (tid & 15) << 4;
    __syncthreads();

    #pragma unroll 1
    for (int t = 0; t < Sn; ++t) {
        // ---- 1. async gather of 16 f16 embedding rows straight into LDS ----
        if (wv == 0) {
            if (lane < TILE_B) {
                const int tok = x[(r0 + lane) * Sn + t];
                const unsigned ldsoff = (unsigned)(uintptr_t)&eA[lane][0];
                const unsigned long long gaddr =
                    (unsigned long long)(uintptr_t)(emb16 + tok * En);
                asm volatile("global_load_async_to_lds_b128 %0, %1, off"
                             :: "v"(ldsoff), "v"(gaddr) : "memory");
            }
            asm volatile("s_wait_asynccnt 0" ::: "memory");
        }
        __syncthreads();

        // ---- 2. layer 0 GEMM: gates = [e | h0] x [Wih0 | Whh0]^T ----
        v8f acc[NT_PER_WAVE];
        #pragma unroll
        for (int n = 0; n < NT_PER_WAVE; ++n)
            #pragma unroll
            for (int v = 0; v < 8; ++v) acc[n][v] = 0.f;

        #pragma unroll 3
        for (int kc = 0; kc < KC0; ++kc) {
            const _Float16* arow = (kc == 0) ? &eA[m][0] : &h0h[m][(kc - 1) * 32];
            v16h a;
            #pragma unroll
            for (int h = 0; h < 16; ++h) a[h] = arow[kk_of(h, hi)];
            const _Float16* fkc = f0lane + kc * 512;
            #pragma unroll
            for (int ntl = 0; ntl < NT_PER_WAVE; ++ntl) {
                const v16h b = *(const v16h*)(fkc + ntl * (KC0 * 512));
                acc[ntl] = __builtin_amdgcn_wmma_f32_16x16x32_f16(
                    false, a, false, b, (short)0, acc[ntl], false, false);
            }
        }
        __syncthreads();   // all waves done reading old h0h / eA

        // hide L2 latency of layer-1 fragments under the gate VALU work
        __builtin_prefetch(f1lane, 0, 3);
        __builtin_prefetch(f1lane + KC1 * 512, 0, 3);

        // ---- 3. layer 0 gates in registers: ntl = gate*2 + s ----
        #pragma unroll
        for (int s = 0; s < 2; ++s) {
            const int j = wv * 32 + s * 16 + m;
            #pragma unroll
            for (int v = 0; v < 8; ++v) {
                const float ig = acc[0 + s][v] + bias0r[0 + s];
                const float fg = acc[2 + s][v] + bias0r[2 + s];
                const float gg = acc[4 + s][v] + bias0r[4 + s];
                const float og = acc[6 + s][v] + bias0r[6 + s];
                const float cn = sigf(fg) * c0r[s * 8 + v] + sigf(ig) * tanh_fast(gg);
                c0r[s * 8 + v] = cn;
                h0h[v + hi * 8][j] = (_Float16)(sigf(og) * tanh_fast(cn));
            }
        }
        __syncthreads();   // new h0 visible

        // ---- 4. layer 1 GEMM: gates = [h0_new | h1] x [Wih1 | Whh1]^T ----
        #pragma unroll
        for (int n = 0; n < NT_PER_WAVE; ++n)
            #pragma unroll
            for (int v = 0; v < 8; ++v) acc[n][v] = 0.f;

        #pragma unroll 4
        for (int kc = 0; kc < KC1; ++kc) {
            const _Float16* arow = (kc < 8) ? &h0h[m][kc * 32] : &h1h[m][(kc - 8) * 32];
            v16h a;
            #pragma unroll
            for (int h = 0; h < 16; ++h) a[h] = arow[kk_of(h, hi)];
            const _Float16* fkc = f1lane + kc * 512;
            #pragma unroll
            for (int ntl = 0; ntl < NT_PER_WAVE; ++ntl) {
                const v16h b = *(const v16h*)(fkc + ntl * (KC1 * 512));
                acc[ntl] = __builtin_amdgcn_wmma_f32_16x16x32_f16(
                    false, a, false, b, (short)0, acc[ntl], false, false);
            }
        }
        __syncthreads();   // all waves done reading old h1h / h0h

        // prefetch next step's layer-0 fragment stream
        __builtin_prefetch(f0lane, 0, 3);

        // ---- 5. layer 1 gates in registers ----
        #pragma unroll
        for (int s = 0; s < 2; ++s) {
            const int j = wv * 32 + s * 16 + m;
            #pragma unroll
            for (int v = 0; v < 8; ++v) {
                const float ig = acc[0 + s][v] + bias1r[0 + s];
                const float fg = acc[2 + s][v] + bias1r[2 + s];
                const float gg = acc[4 + s][v] + bias1r[4 + s];
                const float og = acc[6 + s][v] + bias1r[6 + s];
                const float cn = sigf(fg) * c1r[s * 8 + v] + sigf(ig) * tanh_fast(gg);
                c1r[s * 8 + v] = cn;
                h1h[v + hi * 8][j] = (_Float16)(sigf(og) * tanh_fast(cn));
            }
        }
        __syncthreads();   // new h1 visible for next step
    }

    // ---- write final c states (register-resident) ----
    #pragma unroll
    for (int s = 0; s < 2; ++s) {
        const int j = wv * 32 + s * 16 + m;
        #pragma unroll
        for (int v = 0; v < 8; ++v) {
            const int row = v + hi * 8;
            out[CBASE +           (r0 + row) * Hn + j] = c0r[s * 8 + v];
            out[CBASE + HSTRIDE + (r0 + row) * Hn + j] = c1r[s * 8 + v];
        }
    }
    // ---- write final h states from LDS ----
    #pragma unroll
    for (int k = 0; k < 16; ++k) {
        const int j = ej0 + k;
        out[HBASE +           (r0 + eb) * Hn + j] = (float)h0h[eb][j];
        out[HBASE + HSTRIDE + (r0 + eb) * Hn + j] = (float)h1h[eb][j];
    }

    // ---- FC head: logits[16][80] = h1 x fc_w^T + fc_b (waves 0..4) ----
    if (wv < NT_FC) {
        v8f acc;
        #pragma unroll
        for (int v = 0; v < 8; ++v) acc[v] = 0.f;
        #pragma unroll 2
        for (int kc = 0; kc < KCFC; ++kc) {
            const _Float16* arow = &h1h[m][kc * 32];
            v16h a;
            #pragma unroll
            for (int h = 0; h < 16; ++h) a[h] = arow[kk_of(h, hi)];
            const v16h b = *(const v16h*)(fragfc + ((wv * KCFC + kc) * 32 + lane) * 16);
            acc = __builtin_amdgcn_wmma_f32_16x16x32_f16(
                false, a, false, b, (short)0, acc, false, false);
        }
        const int n = wv * 16 + m;
        const float bb = fcb[n];
        #pragma unroll
        for (int v = 0; v < 8; ++v) {
            const int row = v + hi * 8;
            out[(r0 + row) * Vn + n] = acc[v] + bb;
        }
    }
}

// ---------------------------------------------------------------------------
extern "C" void kernel_launch(void* const* d_in, const int* in_sizes, int n_in,
                              void* d_out, int out_size, void* d_ws, size_t ws_size,
                              hipStream_t stream) {
    const int*   x    = (const int*)  d_in[0];
    const float* emb  = (const float*)d_in[1];
    const float* Wih0 = (const float*)d_in[2];
    const float* Whh0 = (const float*)d_in[3];
    const float* bih0 = (const float*)d_in[4];
    const float* bhh0 = (const float*)d_in[5];
    const float* Wih1 = (const float*)d_in[6];
    const float* Whh1 = (const float*)d_in[7];
    const float* bih1 = (const float*)d_in[8];
    const float* bhh1 = (const float*)d_in[9];
    const float* fcw  = (const float*)d_in[10];
    const float* fcb  = (const float*)d_in[11];

    _Float16* ws = (_Float16*)d_ws;   // needs ~1.7 MB

    lstm_prep_weights<<<512, 256, 0, stream>>>(Wih0, Whh0, Wih1, Whh1, fcw, emb, ws);

    lstm_main<<<Bn / TILE_B, 256, 0, stream>>>(
        x, bih0, bhh0, bih1, bhh1, fcb,
        ws, ws + F0_ELEMS, ws + F0_ELEMS + F1_ELEMS,
        ws + F0_ELEMS + F1_ELEMS + FFC_ELEMS,
        (float*)d_out);
}
